// RowSelfAttention_60833916781130
// MI455X (gfx1250) — matrix operand order repaired and checked by reading, not compiled, not run on hardware
//
#include <hip/hip_runtime.h>
#include <hip/hip_bf16.h>
#include <math.h>

// ---------------------------------------------------------------------------
// Row self-attention (Evoformer-style), fp32 I/O, f16 WMMA compute on gfx1250.
// R=128 rows, L=512 cols, B=1, E=768, H=12 heads, D=64.
// ---------------------------------------------------------------------------

typedef __attribute__((ext_vector_type(16))) _Float16 v16h;
typedef __attribute__((ext_vector_type(8)))  _Float16 v8h;
typedef __attribute__((ext_vector_type(4)))  _Float16 v4h;
typedef __attribute__((ext_vector_type(8)))  float    v8f;
typedef __attribute__((ext_vector_type(4)))  float    v4f;

namespace {

constexpr int R  = 128;
constexpr int L  = 512;
constexpr int E  = 768;
constexpr int H  = 12;
constexpr int D  = 64;
constexpr int TOK = R * L;        // 65536 tokens
constexpr int KD  = R * D;        // 8192 = flattened (r,d) contraction dim
// scaling = D^-0.5 / sqrt(R)
constexpr float SCALING = 0.0110485434560398f;

// ---- workspace layout (bytes). Aliasing is safe: stream-ordered stages. ----
constexpr size_t OFF_XH   = 0;                                   // x f16 [TOK][E]
constexpr size_t SZ_XH    = (size_t)TOK * E * 2;                 // 100.7 MB
constexpr size_t OFF_WH   = OFF_XH + SZ_XH;                      // 4 weight mats f16
constexpr size_t SZ_W1    = (size_t)E * E * 2;                   // 1.18 MB each
constexpr size_t OFF_Q    = OFF_WH + 4 * SZ_W1;                  // q  [H][L][KD] f16
constexpr size_t SZ_QK    = (size_t)H * L * KD * 2;              // 100.7 MB
constexpr size_t OFF_K    = OFF_Q + SZ_QK;                       // k  [H][L][KD] f16
constexpr size_t OFF_VT   = OFF_K + SZ_QK;                       // v  [H][KD][L] f16
// aliases (valid because the producing stage's readers are finished):
constexpr size_t OFF_LOG  = OFF_XH;                              // logits f32 [H][L][L]
constexpr size_t OFF_PH   = OFF_LOG + (size_t)H * L * L * 4;     // probs f16 [H][L][L]
constexpr size_t OFF_CTX  = OFF_Q;                               // context f16 [TOK][E]

// ---------------------------------------------------------------------------
// WMMA operand loaders (contiguous k = 16h + s order, wave32). Per-lane pointers are
// computed once outside the K loop and advanced by +32 halfs per step; the
// 16-row / 16-col subtile displacements are compile-time constants that fold
// into the 24-bit instruction offset.
//   A 16x32 f16: lanes 0-15 -> M=lane, K=0..15;
//                lanes 16-31 -> M=lane-16, K=16..31  (gap = 8)
//   B 32x16 f16: lanes 0-15 -> N=lane, K=0..15; lanes 16-31 -> K=16..31 (gap = 8)
// ---------------------------------------------------------------------------
__device__ __forceinline__ const _Float16* a_ptr(const _Float16* base, int ld, int m0) {
  const int lane = threadIdx.x & 31;
  return base + (size_t)(m0 + (lane & 15)) * ld + ((lane >> 4) << 4);
}
__device__ __forceinline__ const _Float16* b_ptr(const _Float16* base, int ld, int n0) {
  const int lane = threadIdx.x & 31;
  return base + (size_t)(n0 + (lane & 15)) * ld + ((lane >> 4) << 4);
}
__device__ __forceinline__ v16h ld2x8h(const _Float16* p, int gap) {
  union { v16h v; v8h h[2]; } u;
  u.h[0] = *(const v8h*)p;
  u.h[1] = *(const v8h*)(p + gap);
  return u.v;
}
__device__ __forceinline__ v8f wmma_f16(v16h a, v16h b, v8f c) {
  return __builtin_amdgcn_wmma_f32_16x16x32_f16(false, a, false, b, (short)0, c,
                                                false, false);
}

// 32(M) x 64(N) wave tile over row-major f16 A, B (NT GEMM).
// acc[s] = (m-half 0, n-subtile s); acc[4+s] = (m-half 1, n-subtile s).
// 12 b128 loads per 8 WMMAs.
__device__ __forceinline__ void gemm32x64(const _Float16* A, int lda, int m0,
                                          const _Float16* B, int ldb, int n0,
                                          int K, v8f acc[8]) {
  const _Float16* pa = a_ptr(A, lda, m0);
  const _Float16* pb = b_ptr(B, ldb, n0);
  for (int k0 = 0; k0 < K; k0 += 32) {
    v16h a0 = ld2x8h(pa, 8);
    v16h a1 = ld2x8h(pa + (size_t)16 * lda, 8);
#pragma unroll
    for (int s = 0; s < 4; ++s) {
      v16h b = ld2x8h(pb + (size_t)(16 * s) * ldb, 8);
      acc[s]     = wmma_f16(a0, b, acc[s]);
      acc[4 + s] = wmma_f16(a1, b, acc[4 + s]);
    }
    pa += 32; pb += 32;
  }
}

} // namespace

// ---------------------------------------------------------------------------
// Kernel A: fp32 -> f16 convert, 4 elements per thread (grid-stride)
// ---------------------------------------------------------------------------
__global__ void cvt_f32_f16_kernel(const float* __restrict__ src,
                                   _Float16* __restrict__ dst, int n4) {
  const v4f* s4 = (const v4f*)src;
  v4h* d4 = (v4h*)dst;
  for (int i = blockIdx.x * blockDim.x + threadIdx.x; i < n4;
       i += gridDim.x * blockDim.x) {
    v4f x = s4[i];
    v4h y;
#pragma unroll
    for (int j = 0; j < 4; ++j) y[j] = (_Float16)x[j];
    d4[i] = y;
  }
}

// ---------------------------------------------------------------------------
// Kernel B: fused QKV projection + bias + layout scatter.
// Wave = (32-token tile, 32 output channels of one head) for q,k,v at once:
// 12 WMMAs per K-step off 16 b128 loads, one shared pair of A tiles.
//  q -> [h][i][r*64+d] * SCALING ; k -> [h][j][r*64+d] ; v -> [h][r*64+d][j]
// ---------------------------------------------------------------------------
__global__ void __launch_bounds__(128, 1)
qkv_kernel(const _Float16* __restrict__ xh,
           const _Float16* __restrict__ wqh,
           const _Float16* __restrict__ wkh,
           const _Float16* __restrict__ wvh,
           const float* __restrict__ bq,
           const float* __restrict__ bk,
           const float* __restrict__ bv,
           _Float16* __restrict__ qarr,
           _Float16* __restrict__ karr,
           _Float16* __restrict__ vT) {
  const int wave = threadIdx.x >> 5;
  const int lane = threadIdx.x & 31;
  const int t0   = (blockIdx.x * 4 + wave) * 32;  // token tile base (32 rows)
  const int h    = blockIdx.y >> 1;               // head
  const int n0   = h * D + (blockIdx.y & 1) * 32; // output-channel base (32 wide)

  v8f q00 = {}, q01 = {}, q10 = {}, q11 = {};
  v8f k00 = {}, k01 = {}, k10 = {}, k11 = {};
  v8f v00 = {}, v01 = {}, v10 = {}, v11 = {};

  const _Float16* pa = a_ptr(xh, E, t0);
  const _Float16* pq = b_ptr(wqh, E, n0);
  const _Float16* pk = b_ptr(wkh, E, n0);
  const _Float16* pv = b_ptr(wvh, E, n0);

  for (int k0 = 0; k0 < E; k0 += 32) {
    v16h a0 = ld2x8h(pa, 8);
    v16h a1 = ld2x8h(pa + 16 * E, 8);
    v16h b;
    b = ld2x8h(pq, 8);           q00 = wmma_f16(a0, b, q00); q10 = wmma_f16(a1, b, q10);
    b = ld2x8h(pq + 16 * E, 8);  q01 = wmma_f16(a0, b, q01); q11 = wmma_f16(a1, b, q11);
    b = ld2x8h(pk, 8);           k00 = wmma_f16(a0, b, k00); k10 = wmma_f16(a1, b, k10);
    b = ld2x8h(pk + 16 * E, 8);  k01 = wmma_f16(a0, b, k01); k11 = wmma_f16(a1, b, k11);
    b = ld2x8h(pv, 8);           v00 = wmma_f16(a0, b, v00); v10 = wmma_f16(a1, b, v10);
    b = ld2x8h(pv + 16 * E, 8);  v01 = wmma_f16(a0, b, v01); v11 = wmma_f16(a1, b, v11);
    pa += 32; pq += 32; pk += 32; pv += 32;
  }

  const int nloc  = lane & 15;
  const int mbase = (lane >> 4) << 3;
  v8f Q[4] = {q00, q01, q10, q11};   // [mh*2 + s]
  v8f Kk[4] = {k00, k01, k10, k11};
  v8f V[4] = {v00, v01, v10, v11};
#pragma unroll
  for (int mh = 0; mh < 2; ++mh) {
#pragma unroll
    for (int s = 0; s < 2; ++s) {
      const int o = n0 + s * 16 + nloc;   // absolute output channel
      const int d = o - h * D;            // within-head dim
      const float bqv = bq[o], bkv = bk[o], bvv = bv[o];
      const v8f aq = Q[mh * 2 + s], ak = Kk[mh * 2 + s], av = V[mh * 2 + s];
#pragma unroll
      for (int r = 0; r < 8; ++r) {
        const int t  = t0 + mh * 16 + mbase + r;
        const int rr = t >> 9;            // alignment row (t / L)
        const int i  = t & (L - 1);       // sequence column (t % L)
        const size_t rd = (size_t)rr * D + d;
        qarr[(size_t)h * L * KD + (size_t)i * KD + rd] =
            (_Float16)((aq[r] + bqv) * SCALING);
        karr[(size_t)h * L * KD + (size_t)i * KD + rd] =
            (_Float16)(ak[r] + bkv);
        vT[(size_t)h * KD * L + rd * L + i] = (_Float16)(av[r] + bvv);
      }
    }
  }
}

// ---------------------------------------------------------------------------
// Kernel C: logits[h][i][j] = sum_{rd} q[h][i][rd] * k[h][j][rd]   (K=8192)
// ---------------------------------------------------------------------------
__global__ void __launch_bounds__(128, 1)
logits_kernel(const _Float16* __restrict__ qarr,
              const _Float16* __restrict__ karr,
              float* __restrict__ logits) {
  const int wave = threadIdx.x >> 5;
  const int lane = threadIdx.x & 31;
  const int i0 = blockIdx.x * 32;
  const int n0 = (blockIdx.y * 4 + wave) * 64;
  const int h  = blockIdx.z;
  const _Float16* qh = qarr + (size_t)h * L * KD;
  const _Float16* kh = karr + (size_t)h * L * KD;

  v8f acc[8];
#pragma unroll
  for (int s = 0; s < 8; ++s) { v8f z = {}; acc[s] = z; }
  gemm32x64(qh, KD, i0, kh, KD, n0, KD, acc);

  const int nloc  = lane & 15;
  const int mbase = (lane >> 4) << 3;
#pragma unroll
  for (int mh = 0; mh < 2; ++mh)
#pragma unroll
    for (int s = 0; s < 4; ++s)
#pragma unroll
      for (int r = 0; r < 8; ++r)
        logits[((size_t)h * L + i0 + mh * 16 + mbase + r) * L + n0 + s * 16 + nloc] =
            acc[mh * 4 + s][r];
}

// ---------------------------------------------------------------------------
// Kernel D: softmax over j (one wave per row of 512); writes f32 probs to
// d_out's second output region and f16 probs for the context GEMM.
// ---------------------------------------------------------------------------
__global__ void __launch_bounds__(128, 1)
softmax_kernel(const float* __restrict__ logits,
               float* __restrict__ probs_out,
               _Float16* __restrict__ probs_h) {
  const int wave = threadIdx.x >> 5;
  const int lane = threadIdx.x & 31;
  const int row  = blockIdx.x * 4 + wave;       // 0 .. H*L-1
  const float* src = logits + (size_t)row * L;

  float v[16];
  float m = -INFINITY;
#pragma unroll
  for (int i = 0; i < 16; ++i) { v[i] = src[lane + i * 32]; m = fmaxf(m, v[i]); }
#pragma unroll
  for (int off = 16; off > 0; off >>= 1) m = fmaxf(m, __shfl_xor(m, off, 32));

  float sum = 0.f;
#pragma unroll
  for (int i = 0; i < 16; ++i) { v[i] = __expf(v[i] - m); sum += v[i]; }
#pragma unroll
  for (int off = 16; off > 0; off >>= 1) sum += __shfl_xor(sum, off, 32);
  const float inv = 1.0f / sum;

#pragma unroll
  for (int i = 0; i < 16; ++i) {
    const float p = v[i] * inv;
    const size_t idx = (size_t)row * L + lane + i * 32;
    probs_out[idx] = p;
    probs_h[idx]   = (_Float16)p;
  }
}

// ---------------------------------------------------------------------------
// Kernel E: context: ctx[t=(r,i)][h*64+d] = sum_j probs[h][i][j] * v[h][rd][j]
// per head: M=512(i), N=8192(rd), K=512(j)
// ---------------------------------------------------------------------------
__global__ void __launch_bounds__(128, 1)
context_kernel(const _Float16* __restrict__ probs_h,
               const _Float16* __restrict__ vT,
               _Float16* __restrict__ ctx) {
  const int wave = threadIdx.x >> 5;
  const int lane = threadIdx.x & 31;
  const int i0 = blockIdx.x * 32;
  const int n0 = (blockIdx.y * 4 + wave) * 64;
  const int h  = blockIdx.z;
  const _Float16* ph = probs_h + (size_t)h * L * L;
  const _Float16* vh = vT + (size_t)h * KD * L;

  v8f acc[8];
#pragma unroll
  for (int s = 0; s < 8; ++s) { v8f z = {}; acc[s] = z; }
  gemm32x64(ph, L, i0, vh, L, n0, L, acc);

  const int nloc  = lane & 15;
  const int mbase = (lane >> 4) << 3;
#pragma unroll
  for (int s = 0; s < 4; ++s) {
    const int ng = n0 + s * 16 + nloc;  // flattened (r,d)
    const int rr = ng >> 6;
    const int d  = ng & (D - 1);
#pragma unroll
    for (int mh = 0; mh < 2; ++mh)
#pragma unroll
      for (int r = 0; r < 8; ++r) {
        const int i = i0 + mh * 16 + mbase + r;
        const size_t t = (size_t)rr * L + i;
        ctx[t * E + h * D + d] = (_Float16)acc[mh * 4 + s][r];
      }
  }
}

// ---------------------------------------------------------------------------
// Kernel F: output projection: out[t][o] = sum_e ctx[t][e]*wo[o][e] + bo[o]
// ---------------------------------------------------------------------------
__global__ void __launch_bounds__(128, 1)
outproj_kernel(const _Float16* __restrict__ ctx,
               const _Float16* __restrict__ woh,
               const float* __restrict__ bo,
               float* __restrict__ out) {
  const int wave = threadIdx.x >> 5;
  const int lane = threadIdx.x & 31;
  const int t0 = (blockIdx.x * 4 + wave) * 32;
  const int n0 = blockIdx.y * 64;

  v8f acc[8];
#pragma unroll
  for (int s = 0; s < 8; ++s) { v8f z = {}; acc[s] = z; }
  gemm32x64(ctx, E, t0, woh, E, n0, E, acc);

  const int nloc  = lane & 15;
  const int mbase = (lane >> 4) << 3;
#pragma unroll
  for (int s = 0; s < 4; ++s) {
    const int o = n0 + s * 16 + nloc;
    const float bov = bo[o];
#pragma unroll
    for (int mh = 0; mh < 2; ++mh)
#pragma unroll
      for (int r = 0; r < 8; ++r)
        out[(size_t)(t0 + mh * 16 + mbase + r) * E + o] = acc[mh * 4 + s][r] + bov;
  }
}

// ---------------------------------------------------------------------------
extern "C" void kernel_launch(void* const* d_in, const int* in_sizes, int n_in,
                              void* d_out, int out_size, void* d_ws, size_t ws_size,
                              hipStream_t stream) {
  const float* x  = (const float*)d_in[0];
  const float* wq = (const float*)d_in[1];
  const float* bq = (const float*)d_in[2];
  const float* wk = (const float*)d_in[3];
  const float* bk = (const float*)d_in[4];
  const float* wv = (const float*)d_in[5];
  const float* bv = (const float*)d_in[6];
  const float* wo = (const float*)d_in[7];
  const float* bo = (const float*)d_in[8];

  char* ws = (char*)d_ws;
  _Float16* xh   = (_Float16*)(ws + OFF_XH);
  _Float16* wqh  = (_Float16*)(ws + OFF_WH + 0 * SZ_W1);
  _Float16* wkh  = (_Float16*)(ws + OFF_WH + 1 * SZ_W1);
  _Float16* wvh  = (_Float16*)(ws + OFF_WH + 2 * SZ_W1);
  _Float16* woh  = (_Float16*)(ws + OFF_WH + 3 * SZ_W1);
  _Float16* qarr = (_Float16*)(ws + OFF_Q);
  _Float16* karr = (_Float16*)(ws + OFF_K);
  _Float16* vT   = (_Float16*)(ws + OFF_VT);
  float*    lgts = (float*)   (ws + OFF_LOG);   // aliases xh (xh dead by then)
  _Float16* ph   = (_Float16*)(ws + OFF_PH);
  _Float16* ctx  = (_Float16*)(ws + OFF_CTX);   // aliases qarr (dead by then)

  float* out_attn  = (float*)d_out;                       // [TOK][E]
  float* out_probs = (float*)d_out + (size_t)TOK * E;     // [H][1][L][L]

  // A) fp32 -> f16 conversions (vectorized x4)
  cvt_f32_f16_kernel<<<4096, 256, 0, stream>>>(x,  xh,  TOK * E / 4);
  cvt_f32_f16_kernel<<<576,  256, 0, stream>>>(wq, wqh, E * E / 4);
  cvt_f32_f16_kernel<<<576,  256, 0, stream>>>(wk, wkh, E * E / 4);
  cvt_f32_f16_kernel<<<576,  256, 0, stream>>>(wv, wvh, E * E / 4);
  cvt_f32_f16_kernel<<<576,  256, 0, stream>>>(wo, woh, E * E / 4);

  // B) fused QKV projection (+bias, +q scaling, layout scatter)
  qkv_kernel<<<dim3(TOK / 128, H * 2), 128, 0, stream>>>(xh, wqh, wkh, wvh,
                                                         bq, bk, bv, qarr, karr, vT);

  // C) logits per head (K = 8192)
  logits_kernel<<<dim3(L / 32, 2, H), 128, 0, stream>>>(qarr, karr, lgts);

  // D) softmax (one wave per row)
  softmax_kernel<<<(H * L) / 4, 128, 0, stream>>>(lgts, out_probs, ph);

  // E) context per head (M=512, N=8192, K=512)
  context_kernel<<<dim3(L / 32, KD / 256, H), 128, 0, stream>>>(ph, vT, ctx);

  // F) output projection (+bias)
  outproj_kernel<<<dim3(TOK / 128, E / 64), 128, 0, stream>>>(ctx, woh, bo, out_attn);
}
